// DeformableTransformer_13941463843256
// MI455X (gfx1250) — matrix operand order repaired and checked by reading, not compiled, not run on hardware
//
#include <hip/hip_runtime.h>
#include <hip/hip_bf16.h>

// ---------------------------------------------------------------------------
// Multi-Scale Deformable Attention for MI455X (gfx1250, wave32, WMMA + TDM)
//
//   1) off   = query @ W_off  + b_off      GEMM (32768 x 256) x (256 x 256)
//   2) attnl = query @ W_attn + b_attn     GEMM (32768 x 256) x (256 x 128)
//   3) value = input_flatten @ W_val + b   GEMM (87040 x 256) x (256 x 256)
//   4) softmax over 16 (L*P) per (q, head)
//   5) deformable bilinear gather (wave32: lane == HD channel, L2-resident)
//   6) out   = attn_out @ W_out + b_out    GEMM (32768 x 256) x (256 x 256)
//
// GEMM data path:
//   - Weight panel (256x16 fp32, 16KB)  -> LDS via TENSOR_LOAD_TO_LDS (TDM)
//   - A K-tile    (128x32 fp32, 16KB)   -> LDS via GLOBAL_LOAD_ASYNC_TO_LDS_B128
//   - compute: ds_load -> bf16 hi/lo split -> 3x V_WMMA_F32_16X16X32_BF16
// ---------------------------------------------------------------------------

typedef __attribute__((ext_vector_type(16))) __bf16    v16bf;
typedef __attribute__((ext_vector_type(8)))  float     v8f;
typedef __attribute__((ext_vector_type(4)))  unsigned  u32x4;
typedef __attribute__((ext_vector_type(8)))  unsigned  u32x8;

#define N_BATCH 4
#define LQ      8192
#define DM      256
#define MH      8
#define NL      4
#define NP      4
#define HD      32
#define S_TOT   21760
#define KDIM    256

// ---------------------------------------------------------------------------
// TDM: load a (KDIM x 16) fp32 column panel of B (row-major, ld = Nc) to LDS.
// D# group0: count=1, lds_addr, global_addr, type=2
// D# group1: data_size=4B, tensor_dim0=Nc, tensor_dim1=KDIM, tile_dim0=16,
//            tile_dim1=KDIM, tensor_dim0_stride=Nc
// ---------------------------------------------------------------------------
__device__ __forceinline__ void tdm_load_b_panel(const float* gptr,
                                                 unsigned lds_byte_addr,
                                                 int Nc)
{
    const unsigned long long ga = (unsigned long long)(size_t)gptr;
    u32x4 g0;
    g0[0] = 1u;                                   // count=1, user mode
    g0[1] = lds_byte_addr;                        // LDS byte address
    g0[2] = (unsigned)ga;                         // global addr [31:0]
    g0[3] = (unsigned)(ga >> 32) | (2u << 30);    // addr [56:32] | type=2

    u32x8 g1;
    g1[0] = (2u << 16);                           // wg_mask=0, data_size=4B
    g1[1] = ((unsigned)Nc & 0xffffu) << 16;       // tensor_dim0[15:0] @bit48
    g1[2] = ((unsigned)KDIM & 0xffffu) << 16;     // tensor_dim1[15:0] @bit80
    g1[3] = (16u << 16);                          // tile_dim0=16 @bit112
    g1[4] = (unsigned)KDIM;                       // tile_dim1=KDIM @bit128
    g1[5] = (unsigned)Nc;                         // tensor_dim0_stride[31:0]
    g1[6] = 0u;
    g1[7] = 0u;

    asm volatile("tensor_load_to_lds %0, %1" :: "s"(g0), "s"(g1) : "memory");
}

// ---------------------------------------------------------------------------
// GEMM: C[M,Nc] = A[M,256] * B[256,Nc] + bias[Nc]   (row-major fp32)
// Block: 256 threads (8 waves) -> 128(M) x 16(N) output tile.
// Each wave owns one 16x16 tile, K-tile = 32, bf16 hi/lo split (3 WMMA).
// WMMA lane layouts per CDNA5 ISA 7.12.2.
// ---------------------------------------------------------------------------
__global__ __launch_bounds__(256) void msda_gemm_bf16x2(
    const float* __restrict__ A, const float* __restrict__ B,
    const float* __restrict__ bias, float* __restrict__ C,
    int M, int Nc)
{
    __shared__ float sB[KDIM * 16];   // 16 KB: full-K weight panel (TDM)
    __shared__ float sA[128 * 32];    // 16 KB: one K-tile of A (async)

    const int tid    = (int)threadIdx.x;
    const int wid    = tid >> 5;
    const int lane   = tid & 31;
    const int tilesN = Nc >> 4;
    const int tM     = (int)(blockIdx.x / tilesN) << 7;   // 128-row strip
    const int tN     = (int)(blockIdx.x % tilesN) << 4;   // 16-col panel

    // --- stage weight panel with the Tensor Data Mover (wave 0 only) ------
    if (wid == 0) {
        tdm_load_b_panel(B + tN, (unsigned)(size_t)&sB[0], Nc);
        __builtin_amdgcn_s_wait_tensorcnt(0);
    }

    // lane patterns (CDNA5 WMMA 16-bit operand layouts)
    const int laneM = lane & 15;
    const int kA    = (lane >> 4) << 3;     // A k-phase: 0 or 8
    const int kB    = (lane >> 4) << 4;     // unused; B fetch also uses kA (0 or 8)
    const int ncol  = tN + laneM;
    const int arow  = (wid << 4) + laneM;   // row within 128-row strip

    v8f acc = {};
    for (int kt = 0; kt < KDIM; kt += 32) {
        __syncthreads();   // previous compute done before overwriting sA

        // --- async-copy A tile (128 x 32 fp32) global -> LDS --------------
        // 1024 float4s; thread t moves 4 of them (f = i*256 + t).
#pragma unroll
        for (int i = 0; i < 4; ++i) {
            const int f   = i * 256 + tid;
            const int row = f >> 3;          // 0..127
            const int c4  = (f & 7) << 2;    // 0,4,...,28
            const unsigned long long ga =
                (unsigned long long)(size_t)(A + ((long)(tM + row) * KDIM + kt + c4));
            const unsigned loff = (unsigned)(size_t)&sA[row * 32 + c4];
            asm volatile("global_load_async_to_lds_b128 %0, %1, off"
                         :: "v"(loff), "v"(ga) : "memory");
        }
        asm volatile("s_wait_asynccnt 0x0" ::: "memory");
        __syncthreads();   // A tile (and, first pass, TDM B panel) visible

        // --- operand fetch from LDS + bf16 hi/lo split --------------------
        v16bf ahi, alo, bhi, blo;
#pragma unroll
        for (int j = 0; j < 16; ++j) {
            const int kk = kA + (j & 7) + ((j >> 3) << 4);   // 0..31
            const float a = sA[arow * 32 + kk];
            const __bf16 h = (__bf16)a;
            ahi[j] = h;
            alo[j] = (__bf16)(a - (float)h);
        }
#pragma unroll
        for (int j = 0; j < 16; ++j) {
            const float b = sB[(kt + kA + (j & 7) + ((j >> 3) << 4)) * 16 + laneM];
            const __bf16 h = (__bf16)b;
            bhi[j] = h;
            blo[j] = (__bf16)(b - (float)h);
        }

        // A*B ~= Ahi*Bhi + Ahi*Blo + Alo*Bhi  (fp32-class accuracy)
        acc = __builtin_amdgcn_wmma_f32_16x16x32_bf16(
                  false, ahi, false, blo, (short)0, acc, false, false);
        acc = __builtin_amdgcn_wmma_f32_16x16x32_bf16(
                  false, alo, false, bhi, (short)0, acc, false, false);
        acc = __builtin_amdgcn_wmma_f32_16x16x32_bf16(
                  false, ahi, false, bhi, (short)0, acc, false, false);
    }

    // --- store D (f32 16x16 layout: VGPR v -> row (lane>>4)*8+v, col lane&15)
    const float bv = bias[ncol];
    const int rbase = tM + (wid << 4) + ((lane >> 4) << 3);
#pragma unroll
    for (int v = 0; v < 8; ++v) {
        C[(long)(rbase + v) * Nc + ncol] = acc[v] + bv;
    }
}

// ---------------------------------------------------------------------------
// Softmax over contiguous groups of 16 (the L*P axis). One thread per group.
// ---------------------------------------------------------------------------
__global__ __launch_bounds__(256) void msda_softmax16(float* __restrict__ buf,
                                                      int rows)
{
    const int r = (int)(blockIdx.x * blockDim.x + threadIdx.x);
    if (r >= rows) return;
    float* p = buf + (long)r * 16;
    float v[16];
    float mx = -3.0e38f;
#pragma unroll
    for (int i = 0; i < 16; ++i) { v[i] = p[i]; mx = fmaxf(mx, v[i]); }
    float s = 0.f;
#pragma unroll
    for (int i = 0; i < 16; ++i) { v[i] = __expf(v[i] - mx); s += v[i]; }
    const float inv = 1.0f / s;
#pragma unroll
    for (int i = 0; i < 16; ++i) p[i] = v[i] * inv;
}

// ---------------------------------------------------------------------------
// Deformable sampling. One wave per (n, q, m); lane == channel d (HD == 32
// == wave32 width -> every corner read is one coalesced 128B line, served
// mostly from the 192MB L2 since value (89MB) is resident there).
// ---------------------------------------------------------------------------
__global__ __launch_bounds__(256) void msda_sample(
    const float* __restrict__ refpts,  // (N, LQ, NL, 2)
    const float* __restrict__ off,     // (N*LQ, 256) = (m, l, p, 2)
    const float* __restrict__ aw,      // (N*LQ, 128) = (m, l*p) post-softmax
    const float* __restrict__ value,   // (N*S, 256)  = (m, d)
    float* __restrict__ outv)          // (N*LQ, 256) = (m, d)
{
    const int HV[4] = {128, 64, 32, 16};
    const int WV[4] = {128, 64, 32, 16};
    const int ST[4] = {0, 16384, 20480, 21504};

    const int w    = (int)((blockIdx.x * blockDim.x + threadIdx.x) >> 5);
    const int lane = (int)(threadIdx.x & 31);
    const int m = w & (MH - 1);
    const int q = (w >> 3) & (LQ - 1);
    const int n = w >> 16;                 // w / (MH*LQ), MH*LQ = 65536

    const long row = (long)n * LQ + q;
    const float* refp = refpts + row * NL * 2;
    const float* offp = off + row * 256 + m * (NL * NP * 2);
    const float* awp  = aw  + row * 128 + m * (NL * NP);

    float acc = 0.f;
#pragma unroll
    for (int l = 0; l < NL; ++l) {
        const int   H = HV[l], W = WV[l];
        const float fH = (float)H, fW = (float)W;
        const float rx = refp[l * 2 + 0];
        const float ry = refp[l * 2 + 1];
        const long lvlbase = (long)n * S_TOT + ST[l];
#pragma unroll
        for (int p = 0; p < NP; ++p) {
            const float ox = offp[(l * NP + p) * 2 + 0];
            const float oy = offp[(l * NP + p) * 2 + 1];
            const float x = (rx + ox / fW) * fW - 0.5f;
            const float y = (ry + oy / fH) * fH - 0.5f;
            const float wgt = awp[l * NP + p];

            const float x0f = floorf(x), y0f = floorf(y);
            const float lx = x - x0f, ly = y - y0f;
            const int x0 = (int)x0f, y0 = (int)y0f;
            const int x1 = x0 + 1,   y1 = y0 + 1;

            float s = 0.f;
#pragma unroll
            for (int c = 0; c < 4; ++c) {
                const int xi = (c & 1) ? x1 : x0;
                const int yi = (c & 2) ? y1 : y0;
                const float cw = ((c & 1) ? lx : (1.f - lx)) *
                                 ((c & 2) ? ly : (1.f - ly));
                const bool valid = (xi >= 0) & (xi < W) & (yi >= 0) & (yi < H);
                const int xc = min(max(xi, 0), W - 1);
                const int yc = min(max(yi, 0), H - 1);
                const float vv =
                    value[(lvlbase + (long)yc * W + xc) * 256 + m * HD + lane];
                s += vv * (valid ? cw : 0.f);
            }
            acc += wgt * s;
        }
    }
    outv[row * 256 + m * HD + lane] = acc;
}

// ---------------------------------------------------------------------------
// Launch
// ---------------------------------------------------------------------------
extern "C" void kernel_launch(void* const* d_in, const int* in_sizes, int n_in,
                              void* d_out, int out_size, void* d_ws, size_t ws_size,
                              hipStream_t stream)
{
    (void)in_sizes; (void)n_in; (void)out_size; (void)ws_size;

    const float* query  = (const float*)d_in[0];
    const float* refpts = (const float*)d_in[1];
    const float* inflat = (const float*)d_in[2];
    // d_in[3] spatial shapes, d_in[4] level starts: static, baked in.
    const float* W_off  = (const float*)d_in[5];
    const float* b_off  = (const float*)d_in[6];
    const float* W_attn = (const float*)d_in[7];
    const float* b_attn = (const float*)d_in[8];
    const float* W_val  = (const float*)d_in[9];
    const float* b_val  = (const float*)d_in[10];
    const float* W_out  = (const float*)d_in[11];
    const float* b_out  = (const float*)d_in[12];
    float* out = (float*)d_out;

    const long NQ = (long)N_BATCH * LQ;        // 32768 rows
    const long NS = (long)N_BATCH * S_TOT;     // 87040 rows

    float* ws       = (float*)d_ws;
    float* off_buf  = ws;                              // 32768*256
    float* attn_buf = off_buf  + NQ * 256;             // 32768*128
    float* val_buf  = attn_buf + NQ * 128;             // 87040*256
    float* outv_buf = val_buf  + NS * 256;             // 32768*256

    auto gemm = [&](const float* A, const float* B, const float* bias,
                    float* C, int M, int Nc) {
        const int blocks = (M >> 7) * (Nc >> 4);       // 128x16 tile per block
        msda_gemm_bf16x2<<<blocks, 256, 0, stream>>>(A, B, bias, C, M, Nc);
    };

    gemm(query,  W_off,  b_off,  off_buf,  (int)NQ, 256);
    gemm(query,  W_attn, b_attn, attn_buf, (int)NQ, 128);
    gemm(inflat, W_val,  b_val,  val_buf,  (int)NS, 256);

    {
        const int rows = (int)(NQ * MH);               // 262144 groups of 16
        msda_softmax16<<<(rows + 255) / 256, 256, 0, stream>>>(attn_buf, rows);
    }

    {
        const long waves   = NQ * MH;                  // 262144 waves
        const int  blocks  = (int)((waves * 32 + 255) / 256);
        msda_sample<<<blocks, 256, 0, stream>>>(refpts, off_buf, attn_buf,
                                                val_buf, outv_buf);
    }

    gemm(outv_buf, W_out, b_out, out, (int)NQ, 256);
}
